// Block_58892591563088
// MI455X (gfx1250) — matrix operand order repaired and checked by reading, not compiled, not run on hardware
//
#include <hip/hip_runtime.h>
#include <hip/hip_bf16.h>
#include <math.h>
#include <stdint.h>

typedef __attribute__((ext_vector_type(16))) __bf16 v16bf;
typedef __attribute__((ext_vector_type(8)))  __bf16 v8bf;
typedef __attribute__((ext_vector_type(4)))  __bf16 v4bf;
typedef __attribute__((ext_vector_type(8)))  float  v8f;

constexpr int kB  = 4;
constexpr int kS  = 1024;
constexpr int kE  = 1024;
constexpr int kH  = 16;
constexpr int kDH = 64;
constexpr int kM  = kB * kS;   // 4096 token rows

// ---------------------------------------------------------------------------
// WMMA helpers (V_WMMA_F32_16X16X32_BF16, wave32)
// ---------------------------------------------------------------------------
__device__ __forceinline__ v8f wmma_bf16(v16bf a, v16bf b, v8f c) {
  return __builtin_amdgcn_wmma_f32_16x16x32_bf16(
      /*neg_a=*/false, a, /*neg_b=*/false, b,
      /*c_mod=*/(short)0, c, /*reuse_a=*/false, /*reuse_b=*/false);
}

// A-frag (16x32 bf16): lane<16 holds row m, K = 0..15; lane>=16 K = 16..31
__device__ __forceinline__ v16bf load_a_frag(const __bf16* base, int stride,
                                             int m, int k0, int lane) {
  int klo = (lane < 16) ? 0 : 16;
  const __bf16* p = base + (size_t)m * stride + k0 + klo;
  union { v16bf v; v8bf h[2]; } u;
  u.h[0] = *(const v8bf*)p;         // K = klo .. klo+7
  u.h[1] = *(const v8bf*)(p + 8);   // K = klo+8 .. klo+15
  return u.v;
}

// B-frag (32x16 bf16): lane n=L%16; lane<16 K=0..15, lane>=16 K=16..31 (contiguous in K)
__device__ __forceinline__ v16bf load_b_frag(const __bf16* base, int stride,
                                             int n, int k0, int lane) {
  int ks = k0 + ((lane < 16) ? 0 : 16);
  const __bf16* p = base + (size_t)n * stride + ks;
  union { v16bf v; v8bf h[2]; } u;
  u.h[0] = *(const v8bf*)p;
  u.h[1] = *(const v8bf*)(p + 8);
  return u.v;
}

// Async global -> LDS copy, 16B per lane (tracked by ASYNCcnt).
__device__ __forceinline__ void async_ld_b128(uint32_t lds_off, const void* gaddr) {
  asm volatile("global_load_async_to_lds_b128 %0, %1, off"
               :: "v"(lds_off), "v"(gaddr) : "memory");
}
__device__ __forceinline__ uint32_t lds_off_of(const void* p) {
  return (uint32_t)(uintptr_t)p;   // low 32 bits of generic ptr = LDS byte offset
}

// ---------------------------------------------------------------------------
// fp32 -> bf16 conversion (weights / one-shot casts)
// ---------------------------------------------------------------------------
__global__ void f32_to_bf16_kernel(const float* __restrict__ in,
                                   __bf16* __restrict__ out, int n) {
  int i = blockIdx.x * 256 + threadIdx.x;
  if (i < n) out[i] = (__bf16)in[i];
}

// ---------------------------------------------------------------------------
// LayerNorm: one 256-thread block per token row, bf16 output
// ---------------------------------------------------------------------------
__global__ __launch_bounds__(256) void layernorm_bf16_kernel(
    const float* __restrict__ x, const float* __restrict__ g,
    const float* __restrict__ b, __bf16* __restrict__ out) {
  int row  = blockIdx.x;
  int lane = threadIdx.x & 31;
  int wv   = threadIdx.x >> 5;
  const float* xr = x + (size_t)row * kE;
  int i = threadIdx.x * 4;
  float4 xv = *(const float4*)(xr + i);
  float s  = xv.x + xv.y + xv.z + xv.w;
  float ss = xv.x*xv.x + xv.y*xv.y + xv.z*xv.z + xv.w*xv.w;
  #pragma unroll
  for (int off = 1; off < 32; off <<= 1) {
    s  += __shfl_xor(s,  off, 32);
    ss += __shfl_xor(ss, off, 32);
  }
  __shared__ float sm[16];
  if (lane == 0) { sm[wv] = s; sm[8 + wv] = ss; }
  __syncthreads();
  s = 0.f; ss = 0.f;
  #pragma unroll
  for (int w = 0; w < 8; ++w) { s += sm[w]; ss += sm[8 + w]; }
  float mu   = s * (1.0f / kE);
  float var  = ss * (1.0f / kE) - mu * mu;
  float rstd = rsqrtf(var + 1e-5f);
  float4 gv = *(const float4*)(g + i);
  float4 bv = *(const float4*)(b + i);
  v4bf o;
  o[0] = (__bf16)((xv.x - mu) * rstd * gv.x + bv.x);
  o[1] = (__bf16)((xv.y - mu) * rstd * gv.y + bv.y);
  o[2] = (__bf16)((xv.z - mu) * rstd * gv.z + bv.z);
  o[3] = (__bf16)((xv.w - mu) * rstd * gv.w + bv.w);
  *(v4bf*)(out + (size_t)row * kE + i) = o;
}

// ---------------------------------------------------------------------------
// WMMA GEMM with async double-buffered LDS staging.
// Y[m][n] = sum_k A[m][k] * W[n][k] + bias[n]  (+ epilogues)
// Block = 8 waves (2 x 4), wave tile = 32 x 64, block tile = 64 x 256.
// LDS: 2 x (64x32 A-tile = 4KB, 256x32 W-tile = 16KB) = 40KB.
// Each thread stages 5 x 16B per k-step via global_load_async_to_lds_b128.
// EPI 0: split-store bf16 into q/k/v       (QKV projection)
// EPI 1: f32 store out0 = Y + resid         (proj layers + residual)
// EPI 2: bf16 store gelu(Y)                 (MLP up-projection)
// ---------------------------------------------------------------------------
template <int EPI>
__global__ __launch_bounds__(256) void gemm_bf16_kernel(
    const __bf16* __restrict__ A, const __bf16* __restrict__ W,
    const float* __restrict__ bias, const float* __restrict__ resid,
    void* __restrict__ out0, void* __restrict__ out1, void* __restrict__ out2,
    int M, int N, int K) {
  __shared__ __bf16 smA[2][64 * 32];
  __shared__ __bf16 smW[2][256 * 32];

  int tid  = threadIdx.x;
  int lane = tid & 31;
  int wave = tid >> 5;
  int wm = wave & 1, wn = wave >> 1;
  int nl = lane & 15, hf = lane >> 4;
  int blkM = blockIdx.y * 64;
  int blkN = blockIdx.x * 256;

  auto stage = [&](int buf, int k0) {
    {   // A tile: 64x32 bf16 = 256 x 16B chunks, one per thread
      int row = tid >> 2, col = (tid & 3) * 8;
      async_ld_b128(lds_off_of(&smA[buf][row * 32 + col]),
                    A + (size_t)(blkM + row) * K + k0 + col);
    }
    #pragma unroll
    for (int i = 0; i < 4; ++i) {  // W tile: 256x32 bf16 = 1024 chunks
      int c = tid + i * 256;
      int row = c >> 2, col = (c & 3) * 8;
      async_ld_b128(lds_off_of(&smW[buf][row * 32 + col]),
                    W + (size_t)(blkN + row) * K + k0 + col);
    }
  };

  v8f acc[2][4] = {};
  int nkt = K / 32;
  stage(0, 0);
  for (int kt = 0; kt < nkt; ++kt) {
    int cur = kt & 1;
    if (kt + 1 < nkt) {
      stage(cur ^ 1, (kt + 1) * 32);
      asm volatile("s_wait_asynccnt 5" ::: "memory");  // current tile complete
    } else {
      asm volatile("s_wait_asynccnt 0" ::: "memory");
    }
    __syncthreads();

    v16bf a0 = load_a_frag(&smA[cur][0], 32, wm * 32 + nl,      0, lane);
    v16bf a1 = load_a_frag(&smA[cur][0], 32, wm * 32 + 16 + nl, 0, lane);
    #pragma unroll
    for (int j = 0; j < 4; ++j) {
      v16bf b = load_b_frag(&smW[cur][0], 32, wn * 64 + j * 16 + nl, 0, lane);
      acc[0][j] = wmma_bf16(a0, b, acc[0][j]);
      acc[1][j] = wmma_bf16(a1, b, acc[1][j]);
    }
    __syncthreads();   // all waves done reading before this buffer is restaged
  }

  #pragma unroll
  for (int mi = 0; mi < 2; ++mi) {
    #pragma unroll
    for (int j = 0; j < 4; ++j) {
      int n = blkN + wn * 64 + j * 16 + nl;
      float bv = bias[n];
      #pragma unroll
      for (int r = 0; r < 8; ++r) {
        int m = blkM + wm * 32 + mi * 16 + hf * 8 + r;
        float y = acc[mi][j][r] + bv;
        if (EPI == 0) {
          __bf16* dst; int col = n;
          if (col < kE)          { dst = (__bf16*)out0; }
          else if (col < 2 * kE) { dst = (__bf16*)out1; col -= kE; }
          else                   { dst = (__bf16*)out2; col -= 2 * kE; }
          dst[(size_t)m * kE + col] = (__bf16)y;
        } else if (EPI == 1) {
          ((float*)out0)[(size_t)m * N + n] = y + resid[(size_t)m * N + n];
        } else {
          float t = y + 0.044715f * y * y * y;
          float gel = 0.5f * y * (1.0f + tanhf(0.7978845608028654f * t));
          ((__bf16*)out0)[(size_t)m * N + n] = (__bf16)gel;
        }
      }
    }
  }
}

// ---------------------------------------------------------------------------
// Causal flash-style attention. One wave per (batch b, 16-row query tile),
// looping all heads; two-pass online softmax; WMMA for QK^T and PV.
// LDS: att_weights accumulator 16 x S f32 (64KB) + 16 x 32 f32 P scratch.
// ---------------------------------------------------------------------------
__global__ __launch_bounds__(32) void attention_kernel(
    const __bf16* __restrict__ q, const __bf16* __restrict__ k,
    const __bf16* __restrict__ v, __bf16* __restrict__ att,
    float* __restrict__ attw_out) {
  extern __shared__ float smem[];
  float* att_w = smem;            // 16 * kS
  float* sp    = smem + 16 * kS;  // 16 * 32 P scratch

  int lane = threadIdx.x;
  int nl = lane & 15, hf = lane >> 4;
  int qb = blockIdx.x * 16;
  int b  = blockIdx.y;
  size_t rowbase = (size_t)b * kS;
  const float scale = 0.125f;         // 1/sqrt(64)
  int nt = qb / 16 + 1;               // causal: only tiles with keys <= qb+15

  for (int i = lane; i < 16 * kS / 4; i += 32)
    ((float4*)att_w)[i] = make_float4(0.f, 0.f, 0.f, 0.f);
  __syncthreads();

  for (int h = 0; h < kH; ++h) {
    const __bf16* qh = q + rowbase * kE + h * kDH;
    const __bf16* kh = k + rowbase * kE + h * kDH;
    v16bf qa0 = load_a_frag(qh, kE, qb + nl, 0, lane);
    v16bf qa1 = load_a_frag(qh, kE, qb + nl, 32, lane);

    float mrow[8], lrow[8];
    #pragma unroll
    for (int r = 0; r < 8; ++r) { mrow[r] = -1e30f; lrow[r] = 0.f; }

    // ---- pass 1: row max / sum-exp (online) ----
    for (int kt = 0; kt < nt; ++kt) {
      v8f sacc = {};
      v16bf kb0 = load_b_frag(kh, kE, kt * 16 + nl, 0, lane);
      v16bf kb1 = load_b_frag(kh, kE, kt * 16 + nl, 32, lane);
      sacc = wmma_bf16(qa0, kb0, sacc);
      sacc = wmma_bf16(qa1, kb1, sacc);
      int kk = kt * 16 + nl;
      float srow[8], tmax[8];
      #pragma unroll
      for (int r = 0; r < 8; ++r) {
        int qq = qb + hf * 8 + r;
        float sv = sacc[r] * scale;
        if (kk > qq) sv = -1e30f;
        srow[r] = sv;
        float mx = sv;
        #pragma unroll
        for (int off = 1; off < 16; off <<= 1)
          mx = fmaxf(mx, __shfl_xor(mx, off, 32));
        tmax[r] = mx;
      }
      #pragma unroll
      for (int r = 0; r < 8; ++r) {
        float nm = fmaxf(mrow[r], tmax[r]);
        float e  = __expf(srow[r] - nm);
        float sum = e;
        #pragma unroll
        for (int off = 1; off < 16; off <<= 1)
          sum += __shfl_xor(sum, off, 32);
        lrow[r] = lrow[r] * __expf(mrow[r] - nm) + sum;
        mrow[r] = nm;
      }
    }
    float invl[8];
    #pragma unroll
    for (int r = 0; r < 8; ++r) invl[r] = 1.0f / lrow[r];

    // ---- pass 2: probs, PV via WMMA, att_weights accumulation ----
    v8f oacc[4] = {};
    int npairs = (nt + 1) >> 1;
    for (int pt = 0; pt < npairs; ++pt) {
      #pragma unroll
      for (int t = 0; t < 2; ++t) {
        int kt = pt * 2 + t;
        if (kt < nt) {
          v8f sacc = {};
          v16bf kb0 = load_b_frag(kh, kE, kt * 16 + nl, 0, lane);
          v16bf kb1 = load_b_frag(kh, kE, kt * 16 + nl, 32, lane);
          sacc = wmma_bf16(qa0, kb0, sacc);
          sacc = wmma_bf16(qa1, kb1, sacc);
          int kk = kt * 16 + nl;
          #pragma unroll
          for (int r = 0; r < 8; ++r) {
            int row = hf * 8 + r;
            float sv = sacc[r] * scale;
            if (kk > qb + row) sv = -1e30f;
            float p = __expf(sv - mrow[r]);
            sp[row * 32 + t * 16 + nl] = p;
            att_w[row * kS + kk] += p * invl[r] * (1.0f / kH);
          }
        } else {
          #pragma unroll
          for (int r = 0; r < 8; ++r)
            sp[(hf * 8 + r) * 32 + t * 16 + nl] = 0.f;
        }
      }
      __syncthreads();
      // P A-frag (16x32) from LDS, f32 -> bf16
      v16bf pa;
      int klo = (lane < 16) ? 0 : 16;
      #pragma unroll
      for (int i = 0; i < 8; ++i) pa[i]     = (__bf16)sp[nl * 32 + klo + i];
      #pragma unroll
      for (int i = 0; i < 8; ++i) pa[8 + i] = (__bf16)sp[nl * 32 + klo + 8 + i];
      // V B-frags: B[kk][d] = V[kk][d], strided in kk
      #pragma unroll
      for (int j = 0; j < 4; ++j) {
        v16bf vb;
        int d = h * kDH + j * 16 + nl;
        int kkb = pt * 32 + ((lane < 16) ? 0 : 16);
        #pragma unroll
        for (int i = 0; i < 16; ++i) {
          int kk = kkb + i;
          if (kk >= kS) kk = kS - 1;   // zero-prob tail, clamp for safety
          vb[i] = v[(rowbase + kk) * kE + d];
        }
        oacc[j] = wmma_bf16(pa, vb, oacc[j]);
      }
      __syncthreads();
    }

    // write attention output (bf16) for this head
    #pragma unroll
    for (int j = 0; j < 4; ++j)
      #pragma unroll
      for (int r = 0; r < 8; ++r) {
        int m = qb + hf * 8 + r;
        att[(rowbase + m) * kE + h * kDH + j * 16 + nl] =
            (__bf16)(oacc[j][r] * invl[r]);
      }
  }

  __syncthreads();
  float* dst = attw_out + ((size_t)b * kS + qb) * kS;
  for (int i = lane; i < 16 * kS / 4; i += 32)
    ((float4*)dst)[i] = ((const float4*)att_w)[i];
}

// ---------------------------------------------------------------------------
// Launcher
// ---------------------------------------------------------------------------
extern "C" void kernel_launch(void* const* d_in, const int* in_sizes, int n_in,
                              void* d_out, int out_size, void* d_ws, size_t ws_size,
                              hipStream_t stream) {
  (void)in_sizes; (void)n_in; (void)out_size; (void)ws_size;
  const float* x      = (const float*)d_in[0];
  const float* ln1_g  = (const float*)d_in[2];
  const float* ln1_b  = (const float*)d_in[3];
  const float* ln2_g  = (const float*)d_in[4];
  const float* ln2_b  = (const float*)d_in[5];
  const float* w_in   = (const float*)d_in[6];
  const float* b_in   = (const float*)d_in[7];
  const float* w_out  = (const float*)d_in[8];
  const float* b_out  = (const float*)d_in[9];
  const float* w_fc   = (const float*)d_in[10];
  const float* b_fc   = (const float*)d_in[11];
  const float* w_proj = (const float*)d_in[12];
  const float* b_proj = (const float*)d_in[13];

  char* ws = (char*)d_ws;
  size_t off = 0;
  auto alloc = [&](size_t bytes) {
    char* p = ws + off; off += (bytes + 255) & ~size_t(255); return p;
  };
  __bf16* w_in_bf   = (__bf16*)alloc((size_t)3 * kE * kE * 2);
  __bf16* w_out_bf  = (__bf16*)alloc((size_t)kE * kE * 2);
  __bf16* w_fc_bf   = (__bf16*)alloc((size_t)4 * kE * kE * 2);
  __bf16* w_proj_bf = (__bf16*)alloc((size_t)4 * kE * kE * 2);
  __bf16* h_bf      = (__bf16*)alloc((size_t)kM * kE * 2);
  __bf16* q_bf      = (__bf16*)alloc((size_t)kM * kE * 2);
  __bf16* k_bf      = (__bf16*)alloc((size_t)kM * kE * 2);
  __bf16* v_bf      = (__bf16*)alloc((size_t)kM * kE * 2);
  __bf16* att_bf    = (__bf16*)alloc((size_t)kM * kE * 2);
  __bf16* fc_bf     = (__bf16*)alloc((size_t)kM * 4 * kE * 2);
  float*  x1        = (float*)alloc((size_t)kM * kE * 4);

  float* out_x   = (float*)d_out;
  float* out_att = out_x + (size_t)kM * kE;

  auto cvt = [&](const float* src, __bf16* dst, int n) {
    f32_to_bf16_kernel<<<(n + 255) / 256, 256, 0, stream>>>(src, dst, n);
  };
  cvt(w_in,   w_in_bf,   3 * kE * kE);
  cvt(w_out,  w_out_bf,  kE * kE);
  cvt(w_fc,   w_fc_bf,   4 * kE * kE);
  cvt(w_proj, w_proj_bf, 4 * kE * kE);

  // LN1
  layernorm_bf16_kernel<<<kM, 256, 0, stream>>>(x, ln1_g, ln1_b, h_bf);

  // QKV projection (M=4096, N=3072, K=1024)
  gemm_bf16_kernel<0><<<dim3(3 * kE / 256, kM / 64), 256, 0, stream>>>(
      h_bf, w_in_bf, b_in, nullptr, q_bf, k_bf, v_bf, kM, 3 * kE, kE);

  // Attention (one wave per 16-row query tile)
  size_t smem = (size_t)16 * kS * 4 + (size_t)16 * 32 * 4;
  hipFuncSetAttribute((const void*)attention_kernel,
                      hipFuncAttributeMaxDynamicSharedMemorySize, (int)smem);
  attention_kernel<<<dim3(kS / 16, kB), 32, smem, stream>>>(
      q_bf, k_bf, v_bf, att_bf, out_att);

  // Output projection + residual: x1 = x + att @ w_out.T + b_out
  gemm_bf16_kernel<1><<<dim3(kE / 256, kM / 64), 256, 0, stream>>>(
      att_bf, w_out_bf, b_out, x, x1, nullptr, nullptr, kM, kE, kE);

  // LN2
  layernorm_bf16_kernel<<<kM, 256, 0, stream>>>(x1, ln2_g, ln2_b, h_bf);

  // MLP up: fc = gelu(h2 @ w_fc.T + b_fc)  (N=4096)
  gemm_bf16_kernel<2><<<dim3(4 * kE / 256, kM / 64), 256, 0, stream>>>(
      h_bf, w_fc_bf, b_fc, nullptr, fc_bf, nullptr, nullptr, kM, 4 * kE, kE);

  // MLP down + residual: out = x1 + fc @ w_proj.T + b_proj  (K=4096)
  gemm_bf16_kernel<1><<<dim3(kE / 256, kM / 64), 256, 0, stream>>>(
      fc_bf, w_proj_bf, b_proj, x1, out_x, nullptr, nullptr, kM, kE, 4 * kE);
}
